// FMT_39900246179977
// MI455X (gfx1250) — matrix operand order repaired and checked by reading, not compiled, not run on hardware
//
#include <hip/hip_runtime.h>

// ---------------------------------------------------------------------------
// Types for CDNA5 WMMA (wave32)
// ---------------------------------------------------------------------------
typedef __attribute__((ext_vector_type(16))) __bf16 v16bf;
typedef __attribute__((ext_vector_type(8)))  float  v8f;
typedef __attribute__((ext_vector_type(4)))  int    v4i;

#define D_MODEL 256
#define D_FF    512
#define SEQ     4096
#define NBATCH  8              // 2*N after concat
#define MROWS   (NBATCH * SEQ) // 32768
#define NHEAD   8
#define DHEAD   32

#define OP_NONE 0
#define OP_PHI  1   // elu(x)+1
#define OP_RELU 2

#ifndef __has_builtin
#define __has_builtin(x) 0
#endif
#if __has_builtin(__builtin_amdgcn_global_load_async_to_lds_b128)
#define ASYNC_BUILTIN 1
#else
#define ASYNC_BUILTIN 0
#endif
#if __has_builtin(__builtin_amdgcn_s_wait_asynccnt)
#define WAIT_ASYNC_BUILTIN 1
#else
#define WAIT_ASYNC_BUILTIN 0
#endif

#define AS1 __attribute__((address_space(1)))
#define AS3 __attribute__((address_space(3)))

// 16-byte async copy global -> LDS (ASYNCcnt-tracked, no VGPR staging)
__device__ inline void async_cp16(const __bf16* g, __bf16* l) {
#if ASYNC_BUILTIN
    __builtin_amdgcn_global_load_async_to_lds_b128(
        (AS1 v4i*)(g), (AS3 v4i*)(l), 0, 0);
#else
    unsigned lds_off = (unsigned)(uintptr_t)(AS3 __bf16*)(l);
    unsigned long long ga = (unsigned long long)(uintptr_t)g;
    asm volatile("global_load_async_to_lds_b128 %0, %1, off"
                 :: "v"(lds_off), "v"(ga) : "memory");
#endif
}

template<int N>
__device__ inline void wait_async() {
#if WAIT_ASYNC_BUILTIN
    __builtin_amdgcn_s_wait_asynccnt((unsigned short)N);
#else
    asm volatile("s_wait_asynccnt %0" :: "i"(N) : "memory");
#endif
}

// ---------------------------------------------------------------------------
// Positional encoding + NCHW -> (N, L, C) + concat(f0,f1)/(f1,f0)
// ---------------------------------------------------------------------------
__global__ void pos_encode_kernel(const float* __restrict__ ref,
                                  const float* __restrict__ src,
                                  float* __restrict__ x,
                                  __bf16* __restrict__ xb,
                                  __bf16* __restrict__ p2b) {
    size_t gid = (size_t)blockIdx.x * 256 + threadIdx.x; // over 8*4096*256
    int c  = (int)(gid & 255);
    size_t rest = gid >> 8;
    int l  = (int)(rest & 4095);
    int n2 = (int)(rest >> 12);
    int hh = l >> 6, ww = l & 63;

    int i = c >> 2, r = c & 3;
    float div = __expf(-(float)i * (2.0f * 9.210340371976184f / 128.0f));
    float pos = (r < 2) ? (float)ww : (float)hh;
    float arg = pos * div;
    float pe  = (r & 1) ? __cosf(arg) : __sinf(arg);

    const float* sp = (n2 < 4) ? ref : src;
    int nn = n2 & 3;
    float val = sp[(((size_t)nn * D_MODEL + c) * 64 + hh) * 64 + ww] + pe;

    x[gid]  = val;
    xb[gid] = (__bf16)val;
    int n2s = (n2 < 4) ? (n2 + 4) : (n2 - 4);
    p2b[(((size_t)n2s * SEQ + l) << 8) + c] = (__bf16)val;
}

// ---------------------------------------------------------------------------
// Weight transpose f32 (K x N) -> bf16 (N x K)
// ---------------------------------------------------------------------------
__global__ void wtrans_kernel(const float* __restrict__ W,
                              __bf16* __restrict__ WT, int K, int N) {
    int idx = blockIdx.x * 256 + threadIdx.x;
    if (idx >= K * N) return;
    int n = idx / K;
    int k = idx - n * K;
    WT[(size_t)n * K + k] = (__bf16)W[(size_t)k * N + n];
}

// ---------------------------------------------------------------------------
// WMMA GEMM: Y[M,N] = A[M,K](bf16) * B^T[N,K](bf16) + bias, fused epilogue.
// Block tile 128x64, K-step 32, double-buffered async-to-LDS staging.
// 8 waves: 4 M-waves x 2 N-waves, each wave a 2x2 grid of 16x16 WMMA tiles.
// ---------------------------------------------------------------------------
__device__ inline v16bf load_a_frag(const __bf16* rowp, int half) {
    union { uint4 u[2]; v16bf v; } t;
    const uint4* p = reinterpret_cast<const uint4*>(rowp);
    t.u[0] = p[half];       // K = half*8 .. +7
    t.u[1] = p[2 + half];   // K = 16+half*8 .. +7
    return t.v;
}
__device__ inline v16bf load_b_frag(const __bf16* rowp, int half) {
    union { uint4 u[2]; v16bf v; } t;
    const uint4* p = reinterpret_cast<const uint4*>(rowp);
    t.u[0] = p[half];         // K = half*8 .. +7
    t.u[1] = p[2 + half];     // K = 16+half*8 .. +7
    return t.v;
}

template<int OP, bool WF, bool WB>
__global__ __launch_bounds__(256)
void gemm_wmma_kernel(const __bf16* __restrict__ A,
                      const __bf16* __restrict__ BT,
                      const float* __restrict__ bias,
                      float* __restrict__ Yf,
                      __bf16* __restrict__ Yb,
                      int M, int N, int K) {
    __shared__ __bf16 As[2][128][40]; // +8 pad: 80B rows, 16B segments
    __shared__ __bf16 Bs[2][64][40];

    const int t     = threadIdx.x;
    const int lane  = t & 31;
    const int wave  = t >> 5;
    const int wm    = wave & 3;
    const int wn    = wave >> 2;
    const int lid   = lane & 15;
    const int half  = lane >> 4;

    const int mBase = blockIdx.x * 128;
    const int nBase = blockIdx.y * 64;

    // per-thread staging coordinates (3 x 16B per tile)
    const int arow0 = t >> 2,         aseg0 = t & 3;
    const int arow1 = (t + 256) >> 2, aseg1 = t & 3;
    const int brow  = t >> 2,         bseg  = t & 3;

    auto stage = [&](int kk, int b) {
        async_cp16(A  + (size_t)(mBase + arow0) * K + kk + aseg0 * 8, &As[b][arow0][aseg0 * 8]);
        async_cp16(A  + (size_t)(mBase + arow1) * K + kk + aseg1 * 8, &As[b][arow1][aseg1 * 8]);
        async_cp16(BT + (size_t)(nBase + brow ) * K + kk + bseg  * 8, &Bs[b][brow ][bseg  * 8]);
    };

    // preload bias (all GEMMs here have bias)
    float bv[2];
    #pragma unroll
    for (int j = 0; j < 2; ++j) bv[j] = bias[nBase + wn * 32 + j * 16 + lid];

    v8f acc[2][2] = {};

    const int S = K >> 5;
    int buf = 0;
    stage(0, 0);
    for (int s = 0; s < S; ++s) {
        if (s + 1 < S) {
            stage((s + 1) << 5, buf ^ 1);
            wait_async<3>();   // own 3 ops of tile s retired (in-order), 3 in flight
        } else {
            wait_async<0>();
        }
        __syncthreads();       // all threads' copies for tile s visible

        v16bf af[2], bf[2];
        #pragma unroll
        for (int i = 0; i < 2; ++i) af[i] = load_a_frag(&As[buf][wm * 32 + i * 16 + lid][0], half);
        #pragma unroll
        for (int j = 0; j < 2; ++j) bf[j] = load_b_frag(&Bs[buf][wn * 32 + j * 16 + lid][0], half);

        #pragma unroll
        for (int i = 0; i < 2; ++i)
            #pragma unroll
            for (int j = 0; j < 2; ++j)
                acc[i][j] = __builtin_amdgcn_wmma_f32_16x16x32_bf16(
                    false, af[i], false, bf[j], (short)0, acc[i][j], false, false);

        __syncthreads();       // done reading buf before tile s+2 overwrites it
        buf ^= 1;
    }

    // Epilogue: VGPR r -> lanes 0-15 M=r, lanes 16-31 M=r+8; N=lid
    #pragma unroll
    for (int i = 0; i < 2; ++i) {
        #pragma unroll
        for (int j = 0; j < 2; ++j) {
            int col = nBase + wn * 32 + j * 16 + lid;
            #pragma unroll
            for (int r = 0; r < 8; ++r) {
                int row = mBase + wm * 32 + i * 16 + half * 8 + r;
                float v = acc[i][j][r] + bv[j];
                if (OP == OP_PHI)  v = (v > 0.0f) ? (v + 1.0f) : __expf(v);
                if (OP == OP_RELU) v = fmaxf(v, 0.0f);
                size_t off = (size_t)row * N + col;
                if (WF) Yf[off] = v;
                if (WB) Yb[off] = (__bf16)v;
            }
        }
    }
}

// ---------------------------------------------------------------------------
// zero scratch (for atomic accumulation)
// ---------------------------------------------------------------------------
__global__ void zero_kernel(float* __restrict__ p, int n) {
    int i = blockIdx.x * 256 + threadIdx.x;
    if (i < n) p[i] = 0.0f;
}

// ---------------------------------------------------------------------------
// KV[n,h][d][m] = sum_s phiK[n,s,h,d] * V[n,s,h,m]; Ksum[n,h][d] = sum_s phiK
// ---------------------------------------------------------------------------
__global__ __launch_bounds__(256)
void kv_kernel(const float* __restrict__ phiK, const float* __restrict__ V,
               float* __restrict__ KV, float* __restrict__ Ksum) {
    int nh = blockIdx.y;
    int n = nh >> 3, h = nh & 7;
    __shared__ float Ks[32][33];
    __shared__ float Vs[32][33];
    int t  = threadIdx.x;
    int d  = t >> 3;
    int mg = t & 7;
    float acc[4] = {0.f, 0.f, 0.f, 0.f};
    float ksum = 0.f;
    int s0 = blockIdx.x * 512;
    for (int st = 0; st < 512; st += 32) {
        #pragma unroll
        for (int q = 0; q < 4; ++q) {
            int idx = t + q * 256;
            int sr = idx >> 5, sc = idx & 31;
            size_t off = ((size_t)n * SEQ + s0 + st + sr) * D_MODEL + h * DHEAD + sc;
            Ks[sr][sc] = phiK[off];
            Vs[sr][sc] = V[off];
        }
        __syncthreads();
        #pragma unroll 8
        for (int s = 0; s < 32; ++s) {
            float kd = Ks[s][d];
            if (mg == 0) ksum += kd;
            #pragma unroll
            for (int j = 0; j < 4; ++j) acc[j] += kd * Vs[s][mg * 4 + j];
        }
        __syncthreads();
    }
    size_t base = (size_t)nh * 1024;
    #pragma unroll
    for (int j = 0; j < 4; ++j) atomicAdd(&KV[base + d * 32 + mg * 4 + j], acc[j]);
    if (mg == 0) atomicAdd(&Ksum[(size_t)nh * 32 + d], ksum);
}

// ---------------------------------------------------------------------------
// attn[n,l,h*32+m] = (phiQ . KV[:,m]) / (phiQ . Ksum + eps)  -> bf16
// ---------------------------------------------------------------------------
__global__ __launch_bounds__(256)
void attn_out_kernel(const float* __restrict__ phiQ, const float* __restrict__ KV,
                     const float* __restrict__ Ksum, __bf16* __restrict__ attnb) {
    int token = blockIdx.x;
    int n = token >> 12;
    __shared__ float qs[256];
    __shared__ float kss[256];
    int t = threadIdx.x;
    qs[t]  = phiQ[(size_t)token * D_MODEL + t];
    kss[t] = Ksum[(size_t)n * D_MODEL + t];
    __syncthreads();
    int h = t >> 5, m = t & 31;
    const float* kv = KV + ((size_t)n * NHEAD + h) * 1024;
    float acc = 0.f, z = 0.f;
    #pragma unroll
    for (int d = 0; d < DHEAD; ++d) {
        float q = qs[h * DHEAD + d];
        acc += q * kv[d * 32 + m];
        z   += q * kss[h * DHEAD + d];
    }
    attnb[(size_t)token * D_MODEL + t] = (__bf16)(acc / (z + 1e-6f));
}

// ---------------------------------------------------------------------------
// x = LayerNorm(x + tmp) * g + b ; writes f32 + bf16 mirror. Wave32 per token.
// ---------------------------------------------------------------------------
__global__ __launch_bounds__(256)
void add_ln_kernel(float* __restrict__ x, const float* __restrict__ tmp,
                   const float* __restrict__ g, const float* __restrict__ b,
                   __bf16* __restrict__ xb) {
    int wave = threadIdx.x >> 5;
    int lane = threadIdx.x & 31;
    size_t token = (size_t)blockIdx.x * 8 + wave;
    float* xr = x + token * D_MODEL;
    const float* tr = tmp + token * D_MODEL;
    float v[8];
    float s = 0.f;
    #pragma unroll
    for (int i = 0; i < 8; ++i) {
        v[i] = xr[lane + i * 32] + tr[lane + i * 32];
        s += v[i];
    }
    #pragma unroll
    for (int o = 16; o > 0; o >>= 1) s += __shfl_xor(s, o, 32);
    float mean = s * (1.0f / 256.0f);
    float vs = 0.f;
    #pragma unroll
    for (int i = 0; i < 8; ++i) { float d = v[i] - mean; vs += d * d; }
    #pragma unroll
    for (int o = 16; o > 0; o >>= 1) vs += __shfl_xor(vs, o, 32);
    float inv = rsqrtf(vs * (1.0f / 256.0f) + 1e-5f);
    #pragma unroll
    for (int i = 0; i < 8; ++i) {
        int c = lane + i * 32;
        float y = (v[i] - mean) * inv * g[c] + b[c];
        xr[c] = y;
        xb[token * D_MODEL + c] = (__bf16)y;
    }
}

// ---------------------------------------------------------------------------
// (N, L, C) -> (N, C, H, W)
// ---------------------------------------------------------------------------
__global__ void final_kernel(const float* __restrict__ x, float* __restrict__ out) {
    size_t gid = (size_t)blockIdx.x * 256 + threadIdx.x;
    int ww = (int)(gid & 63);
    size_t r = gid >> 6;
    int hh = (int)(r & 63); r >>= 6;
    int c  = (int)(r & 255);
    int n  = (int)(r >> 8);
    out[gid] = x[(((size_t)n * SEQ + hh * 64 + ww) << 8) + c];
}

// ---------------------------------------------------------------------------
// Host orchestration
// ---------------------------------------------------------------------------
template<int OP, bool WF, bool WB>
static inline void launch_gemm(const __bf16* A, const __bf16* BT, const float* bias,
                               float* Yf, __bf16* Yb, int M, int N, int K,
                               hipStream_t stream) {
    dim3 grid(M / 128, N / 64);
    gemm_wmma_kernel<OP, WF, WB><<<grid, dim3(256), 0, stream>>>(A, BT, bias, Yf, Yb, M, N, K);
}

extern "C" void kernel_launch(void* const* d_in, const int* in_sizes, int n_in,
                              void* d_out, int out_size, void* d_ws, size_t ws_size,
                              hipStream_t stream) {
    const float* ref = (const float*)d_in[0];
    const float* src = (const float*)d_in[1];
    const float* Wq  = (const float*)d_in[2];
    const float* bq  = (const float*)d_in[3];
    const float* Wk  = (const float*)d_in[4];
    const float* bk  = (const float*)d_in[5];
    const float* Wv  = (const float*)d_in[6];
    const float* bv  = (const float*)d_in[7];
    const float* Wo  = (const float*)d_in[8];
    const float* bo  = (const float*)d_in[9];
    const float* W1  = (const float*)d_in[10];
    const float* b1  = (const float*)d_in[11];
    const float* W2  = (const float*)d_in[12];
    const float* b2  = (const float*)d_in[13];
    const float* g1  = (const float*)d_in[14];
    const float* be1 = (const float*)d_in[15];
    const float* g2  = (const float*)d_in[16];
    const float* be2 = (const float*)d_in[17];

    size_t off = 0;
    char* base = (char*)d_ws;
    auto alloc = [&](size_t bytes) -> char* {
        char* p = base + off;
        off += (bytes + 255) & ~(size_t)255;
        return p;
    };
    const size_t ME = (size_t)MROWS * D_MODEL;
    float*  x     = (float*) alloc(ME * 4);
    __bf16* xb    = (__bf16*)alloc(ME * 2);
    __bf16* p2b   = (__bf16*)alloc(ME * 2);
    float*  tmp   = (float*) alloc(ME * 4);
    float*  phiQ  = (float*) alloc(ME * 4);
    float*  phiK  = (float*) alloc(ME * 4);
    float*  Vf    = (float*) alloc(ME * 4);
    __bf16* attnb = (__bf16*)alloc(ME * 2);
    __bf16* h1b   = (__bf16*)alloc((size_t)MROWS * D_FF * 2);
    float*  KV    = (float*) alloc((size_t)64 * 1056 * 4);
    float*  Ksum  = KV + 64 * 1024;
    __bf16* WT    = (__bf16*)alloc((size_t)4 * 524288 * 2);

    for (int i = 0; i < 4; ++i) {
        __bf16* lb = WT + (size_t)i * 524288;
        wtrans_kernel<<<256, 256, 0, stream>>>(Wq + (size_t)i * 65536,  lb + 0,      256, 256);
        wtrans_kernel<<<256, 256, 0, stream>>>(Wk + (size_t)i * 65536,  lb + 65536,  256, 256);
        wtrans_kernel<<<256, 256, 0, stream>>>(Wv + (size_t)i * 65536,  lb + 131072, 256, 256);
        wtrans_kernel<<<256, 256, 0, stream>>>(Wo + (size_t)i * 65536,  lb + 196608, 256, 256);
        wtrans_kernel<<<512, 256, 0, stream>>>(W1 + (size_t)i * 131072, lb + 262144, 256, 512);
        wtrans_kernel<<<512, 256, 0, stream>>>(W2 + (size_t)i * 131072, lb + 393216, 512, 256);
    }

    pos_encode_kernel<<<ME / 256, 256, 0, stream>>>(ref, src, x, xb, p2b);

    for (int i = 0; i < 4; ++i) {
        const bool is_self = ((i & 1) == 0);   // self, cross, self, cross
        const __bf16* lb = WT + (size_t)i * 524288;
        const __bf16* srcb = is_self ? (const __bf16*)xb : (const __bf16*)p2b;

        launch_gemm<OP_PHI,  true,  false>(xb,   lb + 0,      bq + i * 256, phiQ, nullptr, MROWS, 256, 256, stream);
        launch_gemm<OP_PHI,  true,  false>(srcb, lb + 65536,  bk + i * 256, phiK, nullptr, MROWS, 256, 256, stream);
        launch_gemm<OP_NONE, true,  false>(srcb, lb + 131072, bv + i * 256, Vf,   nullptr, MROWS, 256, 256, stream);

        zero_kernel<<<(64 * 1056 + 255) / 256, 256, 0, stream>>>(KV, 64 * 1056);
        kv_kernel<<<dim3(8, 64), 256, 0, stream>>>(phiK, Vf, KV, Ksum);
        attn_out_kernel<<<MROWS, 256, 0, stream>>>(phiQ, KV, Ksum, attnb);

        launch_gemm<OP_NONE, true,  false>(attnb, lb + 196608, bo + i * 256, tmp, nullptr, MROWS, 256, 256, stream);
        add_ln_kernel<<<MROWS / 8, 256, 0, stream>>>(x, tmp, g1 + i * 256, be1 + i * 256, xb);

        launch_gemm<OP_RELU, false, true >(xb,  lb + 262144, b1 + i * 512, nullptr, h1b, MROWS, 512, 256, stream);
        launch_gemm<OP_NONE, true,  false>(h1b, lb + 393216, b2 + i * 256, tmp, nullptr,  MROWS, 256, 512, stream);
        add_ln_kernel<<<MROWS / 8, 256, 0, stream>>>(x, tmp, g2 + i * 256, be2 + i * 256, xb);
    }

    final_kernel<<<ME / 256, 256, 0, stream>>>(x, (float*)d_out);
}
